// GraphSAGEModel_13477607375484
// MI455X (gfx1250) — matrix operand order repaired and checked by reading, not compiled, not run on hardware
//
#include <hip/hip_runtime.h>
#include <hip/hip_bf16.h>

typedef __bf16 bf16_t;
typedef __bf16 v16bf __attribute__((ext_vector_type(16)));
typedef __bf16 v8bf  __attribute__((ext_vector_type(8)));
typedef float  v8f   __attribute__((ext_vector_type(8)));

#define NNODES 100000
#define NEDGES 1600000
#define DFEAT  128
#define NCLS   40
#define NCPAD  48
#define KCAT   256   // concatenated K: [agg*deg_inv (128) | x (128)]

// ---------------------------------------------------------------- degree ----
__global__ void deg_kernel(const int* __restrict__ dst, float* __restrict__ deg, int E) {
    int t = blockIdx.x * blockDim.x + threadIdx.x;
    if (t < E) atomicAdd(&deg[dst[t]], 1.0f);
}

__global__ void deginv_kernel(float* __restrict__ deg, int N) {
    int t = blockIdx.x * blockDim.x + threadIdx.x;
    if (t < N) deg[t] = 1.0f / fmaxf(deg[t], 1.0f);
}

// ------------------------------------------------- edge gather/scatter-add --
// One wave per edge; each lane moves 4 consecutive floats (float4 gather,
// 4x global_atomic_add_f32 scatter). 128 dims = 32 lanes * 4.
__global__ void agg_kernel(const int* __restrict__ src, const int* __restrict__ dst,
                           const float* __restrict__ feat, float* __restrict__ agg, int E) {
    int t = blockIdx.x * blockDim.x + threadIdx.x;
    int e = t >> 5;
    int lane = t & 31;
    if (e >= E) return;
    int s = src[e];
    int d = dst[e];
    const float4 v = *(const float4*)(feat + (size_t)s * DFEAT + lane * 4);
    float* o = agg + (size_t)d * DFEAT + lane * 4;
    atomicAdd(o + 0, v.x);
    atomicAdd(o + 1, v.y);
    atomicAdd(o + 2, v.z);
    atomicAdd(o + 3, v.w);
}

// ------------------------------------------- prep: scale + concat + ->bf16 --
// abf[n][0:128]   = bf16(agg[n]*deg_inv[n])
// abf[n][128:256] = bf16(f_in[n])
__global__ void prep_kernel(const float* __restrict__ agg, const float* __restrict__ deginv,
                            const float* __restrict__ fin, bf16_t* __restrict__ abf, int N) {
    int t = blockIdx.x * blockDim.x + threadIdx.x;
    if (t >= N * DFEAT) return;
    int n = t >> 7;
    int d = t & 127;
    float di = deginv[n];
    abf[(size_t)n * KCAT + d]         = (bf16_t)(agg[t] * di);
    abf[(size_t)n * KCAT + DFEAT + d] = (bf16_t)fin[t];
}

// ----------------------------- pack weights transposed [N][K] in bf16 -------
// WT[n][k] = k<128 ? Wl[k][n] : Wr[k-128][n]; zero-pad columns n>=nvalid.
__global__ void wcat_kernel(const float* __restrict__ Wl, const float* __restrict__ Wr,
                            bf16_t* __restrict__ WT, int nvalid) {
    int n = blockIdx.x;    // padded column
    int k = threadIdx.x;   // 0..255
    float v = 0.0f;
    if (n < nvalid) v = (k < DFEAT) ? Wl[k * nvalid + n] : Wr[(k - DFEAT) * nvalid + n];
    WT[(size_t)n * KCAT + k] = (bf16_t)v;
}

// ------------------------------------------------------- WMMA GEMM ----------
// out[M x nvalid] = A[M x 256](bf16) @ WT^T + bias (+relu)
// Block = 256 threads = 8 waves; each wave owns one 16-row M-tile and all
// NT n-tiles (A fragment reused across n). WT (NT*16 x 256 bf16) staged in LDS.
template <int NT>
__global__ __launch_bounds__(256)
void gemm_wmma(const bf16_t* __restrict__ A, const bf16_t* __restrict__ WT,
               const float* __restrict__ bias, float* __restrict__ out,
               int M, int nvalid, int ldout, int relu) {
    extern __shared__ bf16_t lds[];
    const int tid = threadIdx.x;

    // cooperative LDS fill of packed transposed weights
    {
        const uint4* wsrc = (const uint4*)WT;
        uint4* wdst = (uint4*)lds;
        const int cnt = NT * 16 * KCAT * 2 / 16;   // 16B chunks
        for (int i = tid; i < cnt; i += 256) wdst[i] = wsrc[i];
    }
    __syncthreads();

    const int lane   = tid & 31;
    const int wave   = tid >> 5;
    const int laneLo = lane & 15;
    const int laneHi = lane >> 4;
    const int m0 = blockIdx.x * 128 + wave * 16;

    v8f acc[NT];
#pragma unroll
    for (int i = 0; i < NT; ++i)
#pragma unroll
        for (int j = 0; j < 8; ++j) acc[i][j] = 0.0f;

    // A fragment base: row M = lane%16 (clamped for ragged tail), K-halves at
    // 8*(lane/16) and 16+8*(lane/16)  -> two contiguous 16B global loads.
    int arow = m0 + laneLo;
    if (arow >= M) arow = M - 1;
    const bf16_t* abase = A + (size_t)arow * KCAT + 8 * laneHi;

#pragma unroll
    for (int kt = 0; kt < KCAT / 32; ++kt) {
        v8bf alo = *(const v8bf*)(abase + kt * 32);
        v8bf ahi = *(const v8bf*)(abase + kt * 32 + 16);
        v16bf afrag;
#pragma unroll
        for (int j = 0; j < 8; ++j) { afrag[j] = alo[j]; afrag[j + 8] = ahi[j]; }

#pragma unroll
        for (int nt = 0; nt < NT; ++nt) {
            // B fragment: col N = lane%16, K-halves at 8*(lane/16) and 16+8*(lane/16)
            const bf16_t* bp = lds + (size_t)(nt * 16 + laneLo) * KCAT + kt * 32 + 8 * laneHi;
            v8bf blo = *(const v8bf*)bp;
            v8bf bhi = *(const v8bf*)(bp + 16);
            v16bf bfrag;
#pragma unroll
            for (int j = 0; j < 8; ++j) { bfrag[j] = blo[j]; bfrag[j + 8] = bhi[j]; }

            acc[nt] = __builtin_amdgcn_wmma_f32_16x16x32_bf16(
                false, afrag, false, bfrag, (short)0, acc[nt], false, false);
        }
    }

    // epilogue: C layout — VGPR r holds row r + 8*(lane/16), col lane%16
#pragma unroll
    for (int nt = 0; nt < NT; ++nt) {
        int col = nt * 16 + laneLo;
        if (col >= nvalid) continue;
        float bv = bias[col];
#pragma unroll
        for (int r = 0; r < 8; ++r) {
            int row = m0 + r + 8 * laneHi;
            if (row < M) {
                float v = acc[nt][r] + bv;
                if (relu) v = fmaxf(v, 0.0f);
                out[(size_t)row * ldout + col] = v;
            }
        }
    }
}

// ------------------------------------------------------- log_softmax --------
__global__ void lsm_kernel(const float* __restrict__ logits, float* __restrict__ out, int N) {
    int n = blockIdx.x * blockDim.x + threadIdx.x;
    if (n >= N) return;
    const float* p = logits + (size_t)n * NCLS;
    float mx = -3.402823466e38f;
    for (int c = 0; c < NCLS; ++c) mx = fmaxf(mx, p[c]);
    float s = 0.0f;
    for (int c = 0; c < NCLS; ++c) s += __expf(p[c] - mx);
    float ls = __logf(s) + mx;
    float* o = out + (size_t)n * NCLS;
    for (int c = 0; c < NCLS; ++c) o[c] = p[c] - ls;
}

// ---------------------------------------------------------------------------
extern "C" void kernel_launch(void* const* d_in, const int* in_sizes, int n_in,
                              void* d_out, int out_size, void* d_ws, size_t ws_size,
                              hipStream_t stream) {
    const float* x   = (const float*)d_in[0];
    const int*   ei  = (const int*)d_in[1];
    const float* W1l = (const float*)d_in[2];
    const float* b1  = (const float*)d_in[3];
    const float* W1r = (const float*)d_in[4];
    const float* W2l = (const float*)d_in[5];
    const float* b2  = (const float*)d_in[6];
    const float* W2r = (const float*)d_in[7];
    const float* W3l = (const float*)d_in[8];
    const float* b3  = (const float*)d_in[9];
    const float* W3r = (const float*)d_in[10];
    float* out = (float*)d_out;

    // workspace carve-out (256B aligned)
    char* ws = (char*)d_ws;
    size_t off = 0;
    auto carve = [&](size_t bytes) -> char* {
        char* p = ws + off;
        off += (bytes + 255) & ~(size_t)255;
        return p;
    };
    float*  deg = (float*)carve((size_t)NNODES * 4);                 // deg -> deg_inv in place
    float*  agg = (float*)carve((size_t)NNODES * DFEAT * 4);
    bf16_t* abf = (bf16_t*)carve((size_t)NNODES * KCAT * 2);
    float*  hA  = (float*)carve((size_t)NNODES * DFEAT * 4);
    float*  hB  = (float*)carve((size_t)NNODES * NCLS * 4);
    bf16_t* wT1 = (bf16_t*)carve((size_t)DFEAT * KCAT * 2);
    bf16_t* wT2 = (bf16_t*)carve((size_t)DFEAT * KCAT * 2);
    bf16_t* wT3 = (bf16_t*)carve((size_t)NCPAD * KCAT * 2);

    const int* src = ei;
    const int* dst = ei + NEDGES;

    const int EB  = (NEDGES + 255) / 256;
    const int NB  = (NNODES + 255) / 256;
    const int AB  = (NEDGES * 32) / 256;              // 32 lanes per edge
    const int PB  = (NNODES * DFEAT + 255) / 256;
    const int GB  = (NNODES + 127) / 128;             // 128 rows per block
    const size_t aggBytes = (size_t)NNODES * DFEAT * 4;

    // degree + inverse
    hipMemsetAsync(deg, 0, (size_t)NNODES * 4, stream);
    deg_kernel<<<EB, 256, 0, stream>>>(dst, deg, NEDGES);
    deginv_kernel<<<NB, 256, 0, stream>>>(deg, NNODES);

    // pack weights (bf16, transposed, K-concatenated)
    wcat_kernel<<<DFEAT, 256, 0, stream>>>(W1l, W1r, wT1, DFEAT);
    wcat_kernel<<<DFEAT, 256, 0, stream>>>(W2l, W2r, wT2, DFEAT);
    wcat_kernel<<<NCPAD, 256, 0, stream>>>(W3l, W3r, wT3, NCLS);

    // ---- layer 1: x -> hA (relu) ----
    hipMemsetAsync(agg, 0, aggBytes, stream);
    agg_kernel<<<AB, 256, 0, stream>>>(src, dst, x, agg, NEDGES);
    prep_kernel<<<PB, 256, 0, stream>>>(agg, deg, x, abf, NNODES);
    gemm_wmma<8><<<GB, 256, DFEAT * KCAT * 2, stream>>>(abf, wT1, b1, hA,
                                                        NNODES, DFEAT, DFEAT, 1);

    // ---- layer 2: hA -> hA (relu) ----
    hipMemsetAsync(agg, 0, aggBytes, stream);
    agg_kernel<<<AB, 256, 0, stream>>>(src, dst, hA, agg, NEDGES);
    prep_kernel<<<PB, 256, 0, stream>>>(agg, deg, hA, abf, NNODES);
    gemm_wmma<8><<<GB, 256, DFEAT * KCAT * 2, stream>>>(abf, wT2, b2, hA,
                                                        NNODES, DFEAT, DFEAT, 1);

    // ---- layer 3: hA -> hB (logits, N=40 padded to 48) ----
    hipMemsetAsync(agg, 0, aggBytes, stream);
    agg_kernel<<<AB, 256, 0, stream>>>(src, dst, hA, agg, NEDGES);
    prep_kernel<<<PB, 256, 0, stream>>>(agg, deg, hA, abf, NNODES);
    gemm_wmma<3><<<GB, 256, NCPAD * KCAT * 2, stream>>>(abf, wT3, b3, hB,
                                                        NNODES, NCLS, NCLS, 0);

    // log_softmax -> output
    lsm_kernel<<<NB, 256, 0, stream>>>(hB, out, NNODES);
}
